// Model_5016521802180
// MI455X (gfx1250) — matrix operand order repaired and checked by reading, not compiled, not run on hardware
//
#include <hip/hip_runtime.h>
#include <hip/hip_bf16.h>

// ---------------------------------------------------------------------------
// Types for CDNA5 WMMA (wave32): v_wmma_f32_16x16x32_bf16
// ---------------------------------------------------------------------------
typedef __attribute__((ext_vector_type(16))) __bf16        bf16x16;
typedef __attribute__((ext_vector_type(8)))  float         f32x8;
typedef __attribute__((ext_vector_type(4)))  unsigned int  u32x4;
typedef __attribute__((ext_vector_type(4)))  int           i32x4;

// async-to-LDS builtin takes int4 pointers in explicit address spaces
typedef __attribute__((address_space(1))) i32x4 gas_i32x4;   // global
typedef __attribute__((address_space(3))) i32x4 las_i32x4;   // LDS

union FragCvt { u32x4 q[2]; bf16x16 v; };

__device__ __forceinline__ unsigned short f2bf(float f) {
    // round-to-nearest-even f32 -> bf16
    unsigned int u = __float_as_uint(f);
    unsigned int r = u + 0x7FFFu + ((u >> 16) & 1u);
    return (unsigned short)(r >> 16);
}

// ---------------------------------------------------------------------------
// Global->LDS 16-byte copy: async-to-LDS when the toolchain exposes it
// (ASYNCcnt path, CDNA5 ISA), else a plain b128 round trip.
// ---------------------------------------------------------------------------
#if __has_builtin(__builtin_amdgcn_global_load_async_to_lds_b128)
#define HAS_ASYNC_LDS 1
#else
#define HAS_ASYNC_LDS 0
#endif

__device__ __forceinline__ void copy16_g2l(unsigned short* lds, const unsigned short* g) {
#if HAS_ASYNC_LDS
    __builtin_amdgcn_global_load_async_to_lds_b128(
        (gas_i32x4*)(void*)g,
        (las_i32x4*)(void*)lds,
        0, 0);
#else
    *(u32x4*)lds = *(const u32x4*)g;
#endif
}

__device__ __forceinline__ void wait_async_copies() {
#if HAS_ASYNC_LDS
#if __has_builtin(__builtin_amdgcn_s_wait_asynccnt)
    __builtin_amdgcn_s_wait_asynccnt(0);
#else
    asm volatile("s_wait_asynccnt 0" ::: "memory");
#endif
#endif
}

// A-fragment (16x32 bf16, MxK): lane<16 holds K {0..7, 16..23}; lane>=16 holds
// K {8..15, 24..31}. rowp points to a 32-element bf16 row in LDS.
__device__ __forceinline__ bf16x16 frag_a(const unsigned short* rowp, int half) {
    FragCvt f;
    f.q[0] = *(const u32x4*)(rowp + half * 8);
    f.q[1] = *(const u32x4*)(rowp + 16 + half * 8);
    return f.v;
}
// B-fragment (32x16 bf16, KxN) from B^T rows stored [n][k]: lane<16 holds
// K {0..7, 16..23}; lane>=16 holds K {8..15, 24..31}.
__device__ __forceinline__ bf16x16 frag_b(const unsigned short* rowp, int half) {
    FragCvt f;
    f.q[0] = *(const u32x4*)(rowp + half * 8);
    f.q[1] = *(const u32x4*)(rowp + 16 + half * 8);
    return f.v;
}

__device__ __forceinline__ float gelu_exact(float x) {
    return 0.5f * x * (1.0f + erff(x * 0.70710678118654752f));
}

// ---------------------------------------------------------------------------
// f32 -> bf16 conversion kernels (run once per operand, bandwidth-trivial)
// ---------------------------------------------------------------------------
__global__ __launch_bounds__(256) void cvt_bf16_kernel(
    const float* __restrict__ in, unsigned short* __restrict__ out, long long n)
{
    long long i4 = ((long long)blockIdx.x * 256 + threadIdx.x) * 4;
    if (i4 + 3 < n) {
        float4 v = *(const float4*)(in + i4);
        unsigned long long pk =
            (unsigned long long)f2bf(v.x) |
            ((unsigned long long)f2bf(v.y) << 16) |
            ((unsigned long long)f2bf(v.z) << 32) |
            ((unsigned long long)f2bf(v.w) << 48);
        *(unsigned long long*)(out + i4) = pk;
    } else {
        for (int j = 0; j < 4; ++j) {
            long long i = i4 + j;
            if (i < n) out[i] = f2bf(in[i]);
        }
    }
}

// in: K x N (row-major), out: N x K (row-major) bf16  (weights -> B^T layout)
__global__ __launch_bounds__(256) void cvt_bf16_t_kernel(
    const float* __restrict__ in, unsigned short* __restrict__ out, int K, int N)
{
    long long gid = (long long)blockIdx.x * 256 + threadIdx.x;
    if (gid >= (long long)N * K) return;
    int k = (int)(gid % K);
    int n = (int)(gid / K);
    out[gid] = f2bf(in[(long long)k * N + n]);
}

// ---------------------------------------------------------------------------
// Generic batched GEMM: C[b] = act(A[b] (MxK, bf16) * Bt (NxK, bf16)^T + bias)
// block = 256 threads = 8 waves; tile 128(M) x 64(N); K-step 32; bf16 WMMA.
// M % 128 == 0 and K % 32 == 0 guaranteed by call sites; N guarded.
// ---------------------------------------------------------------------------
#define GT_M 128
#define GT_N 64
#define GT_K 32

__global__ __launch_bounds__(256) void gemm_wmma_bf16(
    const unsigned short* __restrict__ A, const unsigned short* __restrict__ Bt,
    const float* __restrict__ bias, float* __restrict__ C,
    int M, int N, int K,
    long long sAb, long long sCb, int act)
{
    __shared__ __align__(16) unsigned short As[GT_M][GT_K]; // 8 KB
    __shared__ __align__(16) unsigned short Bs[GT_N][GT_K]; // 4 KB

    A += (long long)blockIdx.z * sAb;
    C += (long long)blockIdx.z * sCb;

    const int m0   = blockIdx.y * GT_M;
    const int n0   = blockIdx.x * GT_N;
    const int tid  = threadIdx.x;
    const int wave = tid >> 5;
    const int lane = tid & 31;
    const int half = lane >> 4;
    const int l16  = lane & 15;

    // per-thread 16B chunk coordinates (4 chunks of 8 bf16 per 32-wide row)
    const int bRow = tid >> 2;      // B tile row (0..63)
    const int bSeg = tid & 3;       // 8-elt segment
    const int gnB  = n0 + bRow;

    f32x8 acc[4] = {};

    for (int k0 = 0; k0 < K; k0 += GT_K) {
        __syncthreads();
        // A tile: 128x32 bf16 = 512 chunks of 16B; 2 per thread
        for (int c = tid; c < (GT_M * GT_K) / 8; c += 256) {
            int r = c >> 2, sg = c & 3;
            copy16_g2l(&As[r][sg * 8], A + (long long)(m0 + r) * K + k0 + sg * 8);
        }
        // B tile: 64x32 bf16 = 256 chunks; 1 per thread (zero-fill OOB rows)
        if (gnB < N) {
            copy16_g2l(&Bs[bRow][bSeg * 8], Bt + (long long)gnB * K + k0 + bSeg * 8);
        } else {
            u32x4 z = {};
            *(u32x4*)&Bs[bRow][bSeg * 8] = z;
        }
        if (k0 + GT_K < K) {
            __builtin_prefetch(A + (long long)(m0 + (tid >> 1)) * K + k0 + GT_K, 0, 1);
        }
        wait_async_copies();
        __syncthreads();

        bf16x16 afrag = frag_a(&As[wave * 16 + l16][0], half);
#pragma unroll
        for (int t = 0; t < 4; ++t) {
            bf16x16 bfrag = frag_b(&Bs[t * 16 + l16][0], half);
            acc[t] = __builtin_amdgcn_wmma_f32_16x16x32_bf16(
                false, afrag, false, bfrag, (short)0, acc[t], false, false);
        }
    }

    // D layout: lane l, reg r -> M = wave*16 + r + 8*(l>>4), N = t*16 + (l&15)
#pragma unroll
    for (int t = 0; t < 4; ++t) {
        int gn = n0 + t * 16 + l16;
        if (gn >= N) continue;
        float bv = bias ? bias[gn] : 0.0f;
#pragma unroll
        for (int r = 0; r < 8; ++r) {
            int gm = m0 + wave * 16 + r + 8 * half;
            float v = acc[t][r] + bv;
            if (act == 1) v = gelu_exact(v);
            C[(long long)gm * N + gn] = v;
        }
    }
}

// ---------------------------------------------------------------------------
// Per-(b,n) mean / stdev over T (population variance, + EPS)
// ---------------------------------------------------------------------------
__global__ __launch_bounds__(256) void stats_kernel(
    const float* __restrict__ xe, float* __restrict__ mean,
    float* __restrict__ stdev, int B, int T, int NV)
{
    int gid = blockIdx.x * 256 + threadIdx.x;
    if (gid >= B * NV) return;
    int b = gid / NV, n = gid % NV;
    const float* p = xe + (long long)b * T * NV + n;
    float s = 0.f, s2 = 0.f;
    for (int t = 0; t < T; ++t) {
        float v = p[(long long)t * NV];
        s += v; s2 += v * v;
    }
    float m  = s / (float)T;
    float va = s2 / (float)T - m * m;
    if (va < 0.f) va = 0.f;
    mean[gid]  = m;
    stdev[gid] = sqrtf(va + 1e-5f);
}

// xnT[(b*NV+n)*T + t] = (xe[b,t,n] - mean) / stdev
__global__ __launch_bounds__(256) void norm_transpose_kernel(
    const float* __restrict__ xe, const float* __restrict__ mean,
    const float* __restrict__ stdev, float* __restrict__ xnT,
    int B, int T, int NV)
{
    long long gid = (long long)blockIdx.x * 256 + threadIdx.x;
    long long total = (long long)B * NV * T;
    if (gid >= total) return;
    int t = (int)(gid % T);
    long long bn = gid / T;
    int n = (int)(bn % NV);
    int b = (int)(bn / NV);
    xnT[gid] = (xe[((long long)b * T + t) * NV + n] - mean[bn]) / stdev[bn];
}

// ---------------------------------------------------------------------------
// SWT decompose: x (rows x D) -> z (rows x 4 x D)
// z[:,0]=approx(level3), z[:,1]=det(d=4), z[:,2]=det(d=2), z[:,3]=det(d=1)
// ---------------------------------------------------------------------------
__global__ __launch_bounds__(256) void swt_dec_kernel(
    const float* __restrict__ x, const float* __restrict__ h0,
    const float* __restrict__ h1, float* __restrict__ z, int NCH, int D)
{
    __shared__ float bufA[512];
    __shared__ float bufB[512];
    float* cur = bufA;
    float* nxt = bufB;
    int row = blockIdx.x;
    int n = row % NCH;
    float w0[4], w1[4];
#pragma unroll
    for (int k = 0; k < 4; ++k) { w0[k] = h0[n * 4 + k]; w1[k] = h1[n * 4 + k]; }
    for (int i = threadIdx.x; i < D; i += 256) cur[i] = x[(long long)row * D + i];
    __syncthreads();
    int dil = 1;
    for (int lev = 0; lev < 3; ++lev) {
        for (int i = threadIdx.x; i < D; i += 256) {
            float det = 0.f, app = 0.f;
#pragma unroll
            for (int k = 0; k < 4; ++k) {
                int j = (i + (k - 1) * dil + D) % D;
                float xv = cur[j];
                det += xv * w1[k];
                app += xv * w0[k];
            }
            z[((long long)row * 4 + (3 - lev)) * D + i] = det;
            nxt[i] = app;
        }
        __syncthreads();
        float* tp = cur; cur = nxt; nxt = tp;
        dil <<= 1;
    }
    for (int i = threadIdx.x; i < D; i += 256)
        z[(long long)row * 4 * D + i] = cur[i];
}

// SWT reconstruct: o (rows x 4 x D) -> y (rows x D)
__global__ __launch_bounds__(256) void swt_rec_kernel(
    const float* __restrict__ o, const float* __restrict__ g0,
    const float* __restrict__ g1, float* __restrict__ y, int NCH, int D)
{
    __shared__ float bufA[512];
    __shared__ float bufB[512];
    float* cur = bufA;
    float* nxt = bufB;
    int row = blockIdx.x;
    int n = row % NCH;
    float w0[4], w1[4];
#pragma unroll
    for (int k = 0; k < 4; ++k) { w0[k] = g0[n * 4 + k]; w1[k] = g1[n * 4 + k]; }
    for (int i = threadIdx.x; i < D; i += 256)
        cur[i] = o[(long long)row * 4 * D + i];
    __syncthreads();
    int dil = 4;
    for (int lev = 0; lev < 3; ++lev) {
        const float* det = o + ((long long)row * 4 + 1 + lev) * D;
        for (int i = threadIdx.x; i < D; i += 256) {
            float s = 0.f;
#pragma unroll
            for (int k = 0; k < 4; ++k) {
                int j = (i + (k - 2) * dil + 2 * D) % D;
                s += cur[j] * w0[k] + det[j] * w1[k];
            }
            nxt[i] = 0.5f * s;
        }
        __syncthreads();
        float* tp = cur; cur = nxt; nxt = tp;
        dil >>= 1;
    }
    for (int i = threadIdx.x; i < D; i += 256)
        y[(long long)row * D + i] = cur[i];
}

// ---------------------------------------------------------------------------
// LayerNorm (optional fused residual): out = LN(a + res) * g + b, row-wise
// ---------------------------------------------------------------------------
__global__ __launch_bounds__(256) void ln_kernel(
    const float* __restrict__ a, const float* __restrict__ res,
    const float* __restrict__ g, const float* __restrict__ be,
    float* __restrict__ out, int D)
{
    __shared__ float buf[512];
    __shared__ float red[256];
    int row = blockIdx.x;
    int tid = threadIdx.x;
    const float* ar = a + (long long)row * D;
    const float* rr = res ? res + (long long)row * D : nullptr;
    float s = 0.f;
    for (int i = tid; i < D; i += 256) {
        float v = ar[i] + (rr ? rr[i] : 0.f);
        buf[i] = v;
        s += v;
    }
    red[tid] = s;
    __syncthreads();
    for (int off = 128; off > 0; off >>= 1) {
        if (tid < off) red[tid] += red[tid + off];
        __syncthreads();
    }
    float mean = red[0] / (float)D;
    __syncthreads();
    float s2 = 0.f;
    for (int i = tid; i < D; i += 256) {
        float d = buf[i] - mean;
        s2 += d * d;
    }
    red[tid] = s2;
    __syncthreads();
    for (int off = 128; off > 0; off >>= 1) {
        if (tid < off) red[tid] += red[tid + off];
        __syncthreads();
    }
    float rstd = rsqrtf(red[0] / (float)D + 1e-5f);
    for (int i = tid; i < D; i += 256)
        out[(long long)row * D + i] = (buf[i] - mean) * rstd * g[i] + be[i];
}

// ---------------------------------------------------------------------------
// ProbAttention for L=4, H=8, Dh=64, u=2. One thread per (row, head).
// q/k/v/out layout: [(row*4 + l)*512 + h*64 + d]
// ---------------------------------------------------------------------------
__device__ __forceinline__ float dot64(const float* __restrict__ a,
                                       const float* __restrict__ b) {
    float s = 0.f;
#pragma unroll
    for (int d = 0; d < 64; ++d) s += a[d] * b[d];
    return s;
}

__global__ __launch_bounds__(256) void prob_attn_kernel(
    const float* __restrict__ q, const float* __restrict__ k,
    const float* __restrict__ v, const int* __restrict__ idxs,
    float* __restrict__ out, int rows)
{
    int gid = blockIdx.x * 256 + threadIdx.x;
    if (gid >= rows * 8) return;
    int h  = gid & 7;
    int bn = gid >> 3;
    const int Dm = 512, Dh = 64;
    const float* Qb = q + (long long)bn * 4 * Dm + h * Dh;
    const float* Kb = k + (long long)bn * 4 * Dm + h * Dh;
    const float* Vb = v + (long long)bn * 4 * Dm + h * Dh;

    float meas[4];
#pragma unroll
    for (int l = 0; l < 4; ++l) {
        float mx = -1e30f, sm = 0.f;
#pragma unroll
        for (int s = 0; s < 2; ++s) {
            int j = idxs[l * 2 + s];
            float d = dot64(Qb + l * Dm, Kb + j * Dm);
            mx = fmaxf(mx, d);
            sm += d;
        }
        meas[l] = mx - sm * 0.25f;
    }
    int t0 = 0;
    for (int l = 1; l < 4; ++l) if (meas[l] > meas[t0]) t0 = l;
    int t1 = -1;
    for (int l = 0; l < 4; ++l) {
        if (l == t0) continue;
        if (t1 < 0 || meas[l] > meas[t1]) t1 = l;
    }
    int tsel[2] = { t0, t1 };
    float attn[2][4];
#pragma unroll
    for (int jj = 0; jj < 2; ++jj) {
        float sc[4];
        float mx = -1e30f;
#pragma unroll
        for (int s = 0; s < 4; ++s) {
            sc[s] = dot64(Qb + tsel[jj] * Dm, Kb + s * Dm) * 0.125f;
            mx = fmaxf(mx, sc[s]);
        }
        float den = 0.f;
#pragma unroll
        for (int s = 0; s < 4; ++s) { sc[s] = expf(sc[s] - mx); den += sc[s]; }
#pragma unroll
        for (int s = 0; s < 4; ++s) attn[jj][s] = sc[s] / den;
    }
    float* Ob = out + (long long)bn * 4 * Dm + h * Dh;
    for (int d = 0; d < Dh; ++d) {
        float v0 = Vb[0 * Dm + d], v1 = Vb[1 * Dm + d];
        float v2 = Vb[2 * Dm + d], v3 = Vb[3 * Dm + d];
        float vm = 0.25f * (v0 + v1 + v2 + v3);
        float u0 = attn[0][0] * v0 + attn[0][1] * v1 + attn[0][2] * v2 + attn[0][3] * v3;
        float u1 = attn[1][0] * v0 + attn[1][1] * v1 + attn[1][2] * v2 + attn[1][3] * v3;
#pragma unroll
        for (int l = 0; l < 4; ++l) {
            float val = (l == t0) ? u0 : (l == t1) ? u1 : vm;
            Ob[l * Dm + d] = val;
        }
    }
}

// dec[b,p,n] = proj[(b*NV+n)*P + p] * stdev[b,n] + mean[b,n]
__global__ __launch_bounds__(256) void finalize_kernel(
    const float* __restrict__ proj, const float* __restrict__ mean,
    const float* __restrict__ stdev, float* __restrict__ out,
    int B, int NV, int P)
{
    int gid = blockIdx.x * 256 + threadIdx.x;
    int total = B * P * NV;
    if (gid >= total) return;
    int n = gid % NV;
    int rem = gid / NV;
    int p = rem % P;
    int b = rem / P;
    long long bn = (long long)b * NV + n;
    out[gid] = proj[bn * (long long)P + p] * stdev[bn] + mean[bn];
}

// ---------------------------------------------------------------------------
// Host orchestration
// ---------------------------------------------------------------------------
static inline void launch_gemm(const unsigned short* A, const unsigned short* Bt,
                               const float* bias, float* C, int M, int N, int K,
                               long long sA, long long sC, int batch, int act,
                               hipStream_t stream)
{
    dim3 grid((N + GT_N - 1) / GT_N, M / GT_M, batch);
    gemm_wmma_bf16<<<grid, 256, 0, stream>>>(A, Bt, bias, C, M, N, K, sA, sC, act);
}

static inline void launch_cvt(const float* in, unsigned short* out, long long n,
                              hipStream_t stream)
{
    long long blocks = (n + 1023) / 1024;
    cvt_bf16_kernel<<<(unsigned)blocks, 256, 0, stream>>>(in, out, n);
}

static inline void launch_cvtT(const float* in, unsigned short* out, int K, int N,
                               hipStream_t stream)
{
    long long tot = (long long)K * N;
    cvt_bf16_t_kernel<<<(unsigned)((tot + 255) / 256), 256, 0, stream>>>(in, out, K, N);
}

extern "C" void kernel_launch(void* const* d_in, const int* in_sizes, int n_in,
                              void* d_out, int out_size, void* d_ws, size_t ws_size,
                              hipStream_t stream)
{
    (void)in_sizes; (void)n_in; (void)out_size; (void)ws_size;

    const float* x_enc   = (const float*)d_in[0];
    const float* embed_W = (const float*)d_in[1];
    const float* embed_b = (const float*)d_in[2];
    const float* h0      = (const float*)d_in[3];
    const float* h1      = (const float*)d_in[4];
    const float* g0      = (const float*)d_in[5];
    const float* g1      = (const float*)d_in[6];
    const float* Wq      = (const float*)d_in[7];
    const float* bq      = (const float*)d_in[8];
    const float* Wk      = (const float*)d_in[9];
    const float* bk      = (const float*)d_in[10];
    const float* Wv      = (const float*)d_in[11];
    const float* bv      = (const float*)d_in[12];
    const float* Wo      = (const float*)d_in[13];
    const float* bo      = (const float*)d_in[14];
    const float* W1      = (const float*)d_in[15];
    const float* b1      = (const float*)d_in[16];
    const float* W2      = (const float*)d_in[17];
    const float* b2      = (const float*)d_in[18];
    const float* ln1_g   = (const float*)d_in[19];
    const float* ln1_b   = (const float*)d_in[20];
    const float* ln2_g   = (const float*)d_in[21];
    const float* ln2_b   = (const float*)d_in[22];
    const float* lnf_g   = (const float*)d_in[23];
    const float* lnf_b   = (const float*)d_in[24];
    const float* proj_W  = (const float*)d_in[25];
    const float* proj_b  = (const float*)d_in[26];
    const int*   idxs    = (const int*)d_in[27];
    float* outp = (float*)d_out;

    const int B = 16, T = 512, NV = 512, D = 512, DFF = 2048, P = 96, EL = 2;
    const int rows  = B * NV;          // 8192
    const int rowsZ = rows * 4;        // 32768

    // ---- workspace layout: f32 region, then bf16 (ushort) region ----
    float* ws = (float*)d_ws;
    const size_t off_mean = 0;
    const size_t off_std  = off_mean + (size_t)rows;
    const size_t off_xnT  = off_std  + (size_t)rows;                 // rows*T; reused attn_out / final-LN
    const size_t off_x    = off_xnT  + (size_t)rows * T;             // rows*D
    const size_t off_z    = off_x    + (size_t)rows * D;             // rowsZ*D; reused ctx
    const size_t off_q    = off_z    + (size_t)rowsZ * D;            // rowsZ*D; reused o-proj
    const size_t off_k    = off_q    + (size_t)rowsZ * D;            // rowsZ*D; reused ffn_mid / proj_out
    const size_t off_v    = off_k    + (size_t)rowsZ * D;            // rowsZ*D; reused ffn_y
    const size_t off_end  = off_v    + (size_t)rowsZ * D;

    float* w_mean = ws + off_mean;
    float* w_std  = ws + off_std;
    float* w_xnT  = ws + off_xnT;
    float* w_x    = ws + off_x;
    float* w_z    = ws + off_z;
    float* w_q    = ws + off_q;
    float* w_k    = ws + off_k;
    float* w_v    = ws + off_v;

    // bf16 arena: one activation buffer (rowsZ*D) + transposed weights
    unsigned short* wsu    = (unsigned short*)(ws + off_end);
    unsigned short* act_bf = wsu;                                    // 16,777,216 ushorts
    unsigned short* wt     = wsu + (size_t)rowsZ * D;
    const size_t wEmb   = 0;                                         // D x T
    const size_t wLayer = (size_t)D * T;                             // per-layer block start
    const size_t lblk   = 4 * (size_t)D * D + 2 * (size_t)D * DFF;   // q,k,v,o,w1,w2
    const size_t wProj  = wLayer + 2 * lblk;                         // P x D

    // ---- convert all weights to transposed bf16 (B^T = [n][k]) ----
    launch_cvtT(embed_W, wt + wEmb, T, D, stream);
    for (int l = 0; l < EL; ++l) {
        unsigned short* base = wt + wLayer + (size_t)l * lblk;
        launch_cvtT(Wq + (size_t)l * D * D,   base,                             D,   D,   stream);
        launch_cvtT(Wk + (size_t)l * D * D,   base + (size_t)D * D,             D,   D,   stream);
        launch_cvtT(Wv + (size_t)l * D * D,   base + 2 * (size_t)D * D,         D,   D,   stream);
        launch_cvtT(Wo + (size_t)l * D * D,   base + 3 * (size_t)D * D,         D,   D,   stream);
        launch_cvtT(W1 + (size_t)l * D * DFF, base + 4 * (size_t)D * D,         D,   DFF, stream);
        launch_cvtT(W2 + (size_t)l * DFF * D, base + 4 * (size_t)D * D + (size_t)D * DFF, DFF, D, stream);
    }
    launch_cvtT(proj_W, wt + wProj, D, P, stream);

    // ---- 1. stats + normalized transpose ----
    stats_kernel<<<(rows + 255) / 256, 256, 0, stream>>>(x_enc, w_mean, w_std, B, T, NV);
    norm_transpose_kernel<<<(unsigned)(((long long)rows * T + 255) / 256), 256, 0, stream>>>(
        x_enc, w_mean, w_std, w_xnT, B, T, NV);

    // ---- 2. embedding: per-batch (NV x T) @ (T x D) + b -> x ----
    launch_cvt(w_xnT, act_bf, (long long)rows * T, stream);
    launch_gemm(act_bf, wt + wEmb, embed_b, w_x, NV, D, T,
                (long long)NV * T, (long long)NV * D, B, 0, stream);

    // ---- 3. encoder layers ----
    for (int l = 0; l < EL; ++l) {
        const unsigned short* base = wt + wLayer + (size_t)l * lblk;
        const unsigned short* wq = base;
        const unsigned short* wk = base + (size_t)D * D;
        const unsigned short* wv = base + 2 * (size_t)D * D;
        const unsigned short* wo = base + 3 * (size_t)D * D;
        const unsigned short* w1 = base + 4 * (size_t)D * D;
        const unsigned short* w2 = base + 4 * (size_t)D * D + (size_t)D * DFF;

        swt_dec_kernel<<<rows, 256, 0, stream>>>(w_x, h0 + (size_t)l * NV * 4,
                                                 h1 + (size_t)l * NV * 4, w_z, NV, D);

        // Q/K/V projections share one bf16 conversion of z
        launch_cvt(w_z, act_bf, (long long)rowsZ * D, stream);
        launch_gemm(act_bf, wq, bq + (size_t)l * D, w_q, rowsZ, D, D, 0, 0, 1, 0, stream);
        launch_gemm(act_bf, wk, bk + (size_t)l * D, w_k, rowsZ, D, D, 0, 0, 1, 0, stream);
        launch_gemm(act_bf, wv, bv + (size_t)l * D, w_v, rowsZ, D, D, 0, 0, 1, 0, stream);

        prob_attn_kernel<<<(rows * 8 + 255) / 256, 256, 0, stream>>>(
            w_q, w_k, w_v, idxs, w_z, rows);

        launch_cvt(w_z, act_bf, (long long)rowsZ * D, stream);
        launch_gemm(act_bf, wo, bo + (size_t)l * D, w_q, rowsZ, D, D, 0, 0, 1, 0, stream);

        swt_rec_kernel<<<rows, 256, 0, stream>>>(w_q, g0 + (size_t)l * NV * 4,
                                                 g1 + (size_t)l * NV * 4, w_xnT, NV, D);

        ln_kernel<<<rows, 256, 0, stream>>>(w_x, w_xnT,
                                            ln1_g + (size_t)l * D, ln1_b + (size_t)l * D,
                                            w_x, D);

        // FFN
        launch_cvt(w_x, act_bf, (long long)rows * D, stream);
        launch_gemm(act_bf, w1, b1 + (size_t)l * DFF, w_k, rows, DFF, D, 0, 0, 1, 1, stream);
        launch_cvt(w_k, act_bf, (long long)rows * DFF, stream);
        launch_gemm(act_bf, w2, b2 + (size_t)l * D, w_v, rows, D, DFF, 0, 0, 1, 0, stream);

        ln_kernel<<<rows, 256, 0, stream>>>(w_x, w_v,
                                            ln2_g + (size_t)l * D, ln2_b + (size_t)l * D,
                                            w_x, D);
    }

    // ---- 4. final LN, projection, denormalizing transpose ----
    ln_kernel<<<rows, 256, 0, stream>>>(w_x, nullptr, lnf_g, lnf_b, w_xnT, D);
    launch_cvt(w_xnT, act_bf, (long long)rows * D, stream);
    launch_gemm(act_bf, wt + wProj, proj_b, w_k, rows, P, D, 0, 0, 1, 0, stream);
    finalize_kernel<<<(B * P * NV + 255) / 256, 256, 0, stream>>>(
        w_k, w_mean, w_std, outp, B, NV, P);
}
